// Decoder_29781303230470
// MI455X (gfx1250) — compile-verified, matrix operand order checked by reading, not run on hardware
//
#include <hip/hip_runtime.h>

#define BB 128
#define SS 4096
#define EE 128
#define HH 8
#define DD 16
#define NLAYER 3
#define SPLITN 8
#define WPB 4
#define TILE_S 32
#define RSTRIDE 130                      // padded bf16 row stride (bank-conflict avoidance)
#define ROWS_PER_WAVE (SS / SPLITN / WPB)   // 128
#define TILES_PER_WAVE (ROWS_PER_WAVE / TILE_S) // 4
#define NPART SPLITN
#define LN_EPS 1e-5f

typedef __attribute__((ext_vector_type(16))) __bf16 v16bf;
typedef __attribute__((ext_vector_type(8)))  float  v8f;

union V16 { v16bf v; unsigned short s[16]; unsigned int u[8]; };

__device__ inline unsigned short f2us(float f) {
  __bf16 b = (__bf16)f;
  return __builtin_bit_cast(unsigned short, b);
}

// ---------------------------------------------------------------- conv head
__global__ __launch_bounds__(128) void conv_head(
    const float* __restrict__ st,
    const float* __restrict__ c1w, const float* __restrict__ c1b,
    const float* __restrict__ bn1g, const float* __restrict__ bn1b,
    const float* __restrict__ bn1m, const float* __restrict__ bn1v,
    const float* __restrict__ c2w, const float* __restrict__ c2b,
    const float* __restrict__ bn2g, const float* __restrict__ bn2b,
    const float* __restrict__ bn2m, const float* __restrict__ bn2v,
    const float* __restrict__ lmw, const float* __restrict__ lmb,
    float* __restrict__ q)
{
  const int b = blockIdx.x, t = threadIdx.x;
  __shared__ float x0[28 * 37], a1[16 * 37], a2[16 * 37];
  for (int i = t; i < 28 * 37; i += 128) x0[i] = st[b * (28 * 37) + i];
  __syncthreads();
  for (int i = t; i < 16 * 37; i += 128) {
    int o = i / 37, hh = i % 37;
    float s = c1b[o];
    for (int c = 0; c < 28; ++c)
      for (int kh = 0; kh < 3; ++kh) {
        int y = hh + kh - 1;
        if (y >= 0 && y < 37) s += x0[c * 37 + y] * c1w[((o * 28 + c) * 3 + kh) * 3 + 1];
      }
    s = (s - bn1m[o]) * rsqrtf(bn1v[o] + LN_EPS) * bn1g[o] + bn1b[o];
    a1[i] = fmaxf(s, 0.f);
  }
  __syncthreads();
  for (int i = t; i < 16 * 37; i += 128) {
    int o = i / 37, hh = i % 37;
    float s = c2b[o];
    for (int c = 0; c < 16; ++c)
      for (int kh = 0; kh < 3; ++kh) {
        int y = hh + kh - 1;
        if (y >= 0 && y < 37) s += a1[c * 37 + y] * c2w[((o * 16 + c) * 3 + kh) * 3 + 1];
      }
    s = (s - bn2m[o]) * rsqrtf(bn2v[o] + LN_EPS) * bn2g[o] + bn2b[o];
    a2[i] = fmaxf(s, 0.f);
  }
  __syncthreads();
  float s = lmb[t];
  for (int j = 0; j < 16 * 37; ++j) s += a2[j] * lmw[j * EE + t];
  q[b * EE + t] = s;
}

// ------------------------------------------------ per-layer, per-batch prep
// qn=LN3(q); qh=qn@wq (x sqrt(D)); wkq[h,e]=sum_d wk[h,e,d]*qh[h,d];
// wkqg = bf16(wkq * ln1_g) (heads padded to 16 with zeros); cq[h]=sum_e ln1_b*wkq
__global__ __launch_bounds__(128) void prep(
    const float* __restrict__ q,
    const float* __restrict__ ln3_g, const float* __restrict__ ln3_b,
    const float* __restrict__ wq, const float* __restrict__ wk,
    const float* __restrict__ ln1_g, const float* __restrict__ ln1_b,
    unsigned short* __restrict__ wkqg, float* __restrict__ cq, int layer)
{
  const int b = blockIdx.x, t = threadIdx.x;
  __shared__ float qn[EE], qhL[EE], wkqL[HH * EE], red[EE];
  float qv = q[b * EE + t];
  red[t] = qv; __syncthreads();
  for (int s = 64; s > 0; s >>= 1) { if (t < s) red[t] += red[t + s]; __syncthreads(); }
  float mean = red[0] * (1.f / EE); __syncthreads();
  red[t] = (qv - mean) * (qv - mean); __syncthreads();
  for (int s = 64; s > 0; s >>= 1) { if (t < s) red[t] += red[t + s]; __syncthreads(); }
  float rs = rsqrtf(red[0] * (1.f / EE) + LN_EPS); __syncthreads();
  qn[t] = (qv - mean) * rs * ln3_g[layer * EE + t] + ln3_b[layer * EE + t];
  __syncthreads();
  {
    const int h = t >> 4, d = t & 15;
    const float* wqp = wq + ((size_t)(layer * HH + h) * EE) * DD + d;
    float a = 0.f;
    for (int e = 0; e < EE; ++e) a += qn[e] * wqp[e * DD];
    qhL[t] = a * 4.0f;                       // fold the *sqrt(D)=4 score scale
  }
  __syncthreads();
  {
    const float g1 = ln1_g[layer * EE + t];
    for (int h = 0; h < HH; ++h) {
      const float* wkp = wk + ((size_t)(layer * HH + h) * EE + t) * DD;
      float a = 0.f;
      for (int d = 0; d < DD; ++d) a += wkp[d] * qhL[h * DD + d];
      wkqL[h * EE + t] = a;
      wkqg[((size_t)b * 16 + h) * EE + t] = f2us(a * g1);
    }
    for (int h = HH; h < 16; ++h) wkqg[((size_t)b * 16 + h) * EE + t] = 0;
  }
  __syncthreads();
  for (int h = 0; h < HH; ++h) {
    red[t] = ln1_b[layer * EE + t] * wkqL[h * EE + t];
    __syncthreads();
    for (int s = 64; s > 0; s >>= 1) { if (t < s) red[t] += red[t + s]; __syncthreads(); }
    if (t == 0) cq[b * HH + h] = red[0];
    __syncthreads();
  }
}

// --------------------------------------- main streaming flash-softmax pass
__global__ __launch_bounds__(128) void attn_pass(
    const float* __restrict__ enc,
    const unsigned short* __restrict__ wkqg,
    const float* __restrict__ cq,
    float* __restrict__ pm, float* __restrict__ pl, float* __restrict__ pacc)
{
  __shared__ unsigned int smem[8320];        // 4 waves * 32*130 bf16 tile; reused for merge
  const int b    = blockIdx.x >> 3;
  const int sp   = blockIdx.x & (SPLITN - 1);
  const int tid  = threadIdx.x;
  const int wave = tid >> 5;
  const int lane = tid & 31;
  const int lmod = lane & 15;
  const int hiH  = lane >> 4;

  unsigned short* tile = (unsigned short*)smem + wave * (TILE_S * RSTRIDE);

  // B operand of the score GEMM: wkqg[b][h=lmod][K]; K layout per ISA 16-bit 32xK pattern
  v16bf Bw[4];
  {
    const unsigned short* wb = wkqg + ((size_t)b * 16 + lmod) * EE;
#pragma unroll
    for (int kc = 0; kc < 4; ++kc) {
      V16 w;
      const int k0 = kc * 32 + (hiH ? 8 : 0);
      const int k1 = kc * 32 + (hiH ? 24 : 16);
#pragma unroll
      for (int j = 0; j < 8; ++j) { w.s[j] = wb[k0 + j]; w.s[8 + j] = wb[k1 + j]; }
      Bw[kc] = w.v;
    }
  }
  const float cql = (lmod < HH) ? cq[b * HH + lmod] : 0.f;

  float mst = -1e30f, lst = 0.f;
  v8f acc[8] = {};

  const int sBase = sp * (SS / SPLITN) + wave * ROWS_PER_WAVE;

  for (int tI = 0; tI < TILES_PER_WAVE; ++tI) {
    const int s0 = sBase + tI * TILE_S;
    const float* rowp = enc + ((size_t)b * SS + s0 + lane) * EE;   // lane = s-row
    if (tI + 1 < TILES_PER_WAVE) __builtin_prefetch(rowp + TILE_S * EE, 0, 0);

    // pass 1: LN stats over this lane's private row
    float sum = 0.f, sq = 0.f;
    const float4* r4 = (const float4*)rowp;
#pragma unroll 8
    for (int e4 = 0; e4 < EE / 4; ++e4) {
      float4 v = r4[e4];
      sum += v.x + v.y + v.z + v.w;
      sq  += v.x * v.x + v.y * v.y + v.z * v.z + v.w * v.w;
    }
    const float mean = sum * (1.f / EE);
    const float var  = sq * (1.f / EE) - mean * mean;
    const float rs   = rsqrtf(var + LN_EPS);

    // pass 2 (cache-hot reload): normalized bf16 tile -> LDS, row-major padded
    unsigned int* outp = (unsigned int*)(tile + lane * RSTRIDE);
#pragma unroll 8
    for (int e4 = 0; e4 < EE / 4; ++e4) {
      float4 v = r4[e4];
      outp[e4 * 2] = (unsigned int)f2us((v.x - mean) * rs) |
                     ((unsigned int)f2us((v.y - mean) * rs) << 16);
      outp[e4 * 2 + 1] = (unsigned int)f2us((v.z - mean) * rs) |
                         ((unsigned int)f2us((v.w - mean) * rs) << 16);
    }
    asm volatile("s_wait_dscnt 0" ::: "memory");   // wave-private tile: no barrier needed

    // scores: wei(32x16) = XN(32x128) @ WkqgT(128x16), 8x wmma bf16
    float we[16];
#pragma unroll
    for (int mt = 0; mt < 2; ++mt) {
      v8f wacc = {};
#pragma unroll
      for (int kc = 0; kc < 4; ++kc) {
        V16 a;
        const unsigned int* rp =
            (const unsigned int*)(tile + (mt * 16 + lmod) * RSTRIDE);
        const int k0 = (kc * 32 + (hiH ? 8 : 0)) >> 1;
        const int k1 = (kc * 32 + (hiH ? 24 : 16)) >> 1;
#pragma unroll
        for (int t4 = 0; t4 < 4; ++t4) { a.u[t4] = rp[k0 + t4]; a.u[4 + t4] = rp[k1 + t4]; }
        wacc = __builtin_amdgcn_wmma_f32_16x16x32_bf16(false, a.v, false, Bw[kc],
                                                       (short)0, wacc, false, false);
      }
#pragma unroll
      for (int v = 0; v < 8; ++v) we[mt * 8 + v] = wacc[v] + cql;
    }

    // online softmax per head (head = lmod; partner lane = lane^16)
    float tmax = we[0];
#pragma unroll
    for (int k = 1; k < 16; ++k) tmax = fmaxf(tmax, we[k]);
    tmax = fmaxf(tmax, __shfl_xor(tmax, 16));
    const float mNew  = fmaxf(mst, tmax);
    const float scOld = __expf(mst - mNew);
    float ps = 0.f;
#pragma unroll
    for (int k = 0; k < 16; ++k) { we[k] = __expf(we[k] - mNew); ps += we[k]; }
    ps += __shfl_xor(ps, 16);
    lst = lst * scOld + ps;
    mst = mNew;

    v8f scv;
#pragma unroll
    for (int v = 0; v < 8; ++v) scv[v] = __shfl(scOld, v + hiH * 8);
#pragma unroll
    for (int ec = 0; ec < 8; ++ec) acc[ec] *= scv;

    // acc(16h x 128e) += P^T(16x32) @ XN(32x128): D-layout of scores == A-layout here
    V16 ap;
#pragma unroll
    for (int j = 0; j < 8; ++j) { ap.s[j] = f2us(we[j]); ap.s[8 + j] = f2us(we[8 + j]); }
#pragma unroll
    for (int ec = 0; ec < 8; ++ec) {
      V16 bx;
      const int e = ec * 16 + lmod;
#pragma unroll
      for (int j = 0; j < 8; ++j) {
        bx.s[j]     = tile[((hiH ? 8 : 0) + j) * RSTRIDE + e];
        bx.s[8 + j] = tile[((hiH ? 24 : 16) + j) * RSTRIDE + e];
      }
      acc[ec] = __builtin_amdgcn_wmma_f32_16x16x32_bf16(false, ap.v, false, bx.v,
                                                        (short)0, acc[ec], false, false);
    }
  }

  // ---- merge 4 wave partials inside the block, emit one (m,l,acc) partial
  __syncthreads();
  float* mrg = (float*)smem;                 // [0..31]=m, [32..63]=l, [64..4159]=acc, [4160..]=ef
  if (lane < HH) { mrg[wave * 8 + lane] = mst; mrg[32 + wave * 8 + lane] = lst; }
  if (!hiH) {
#pragma unroll
    for (int ec = 0; ec < 8; ++ec)
#pragma unroll
      for (int v = 0; v < 8; ++v)
        mrg[64 + (wave * 8 + v) * EE + ec * 16 + lmod] = acc[ec][v];
  }
  __syncthreads();

  const int part = b * NPART + sp;
  if (tid < HH) {
    float M = mrg[tid];
    for (int w = 1; w < WPB; ++w) M = fmaxf(M, mrg[w * 8 + tid]);
    float Lt = 0.f;
    for (int w = 0; w < WPB; ++w) {
      float f = __expf(mrg[w * 8 + tid] - M);
      mrg[4160 + w * 8 + tid] = f;
      Lt += f * mrg[32 + w * 8 + tid];
    }
    pm[part * 8 + tid] = M;
    pl[part * 8 + tid] = Lt;
  }
  __syncthreads();
  for (int h = 0; h < HH; ++h) {
    float a = 0.f;
    for (int w = 0; w < WPB; ++w)
      a += mrg[64 + (w * 8 + h) * EE + tid] * mrg[4160 + w * 8 + h];
    pacc[((size_t)part * 8 + h) * EE + tid] = a;
  }
}

// ------------------------------- combine splits + per-batch layer epilogue
__global__ __launch_bounds__(128) void epilogue(
    const float* __restrict__ pm, const float* __restrict__ pl, const float* __restrict__ pacc,
    const float* __restrict__ ln2_g, const float* __restrict__ ln2_b,
    const float* __restrict__ wv,
    const float* __restrict__ proj_w, const float* __restrict__ proj_b,
    const float* __restrict__ ln4_g, const float* __restrict__ ln4_b,
    const float* __restrict__ ffn_w1, const float* __restrict__ ffn_b1,
    const float* __restrict__ ffn_w2, const float* __restrict__ ffn_b2,
    float* __restrict__ q, int layer)
{
  const int b = blockIdx.x, t = threadIdx.x;
  __shared__ float ef[NPART * HH], invL[HH];
  __shared__ float vbar[HH * EE], attL[EE], qbuf[EE], x4[EE], hbuf[4 * EE], red[EE];

  if (t < HH) {
    float M = -1e30f;
    for (int p = 0; p < NPART; ++p) M = fmaxf(M, pm[(b * NPART + p) * 8 + t]);
    float Lt = 0.f;
    for (int p = 0; p < NPART; ++p) {
      float f = __expf(pm[(b * NPART + p) * 8 + t] - M);
      ef[p * 8 + t] = f;
      Lt += f * pl[(b * NPART + p) * 8 + t];
    }
    invL[t] = 1.f / Lt;
  }
  __syncthreads();
  {
    const float g2 = ln2_g[layer * EE + t], b2 = ln2_b[layer * EE + t];
    for (int h = 0; h < HH; ++h) {
      float a = 0.f;
      for (int p = 0; p < NPART; ++p)
        a += pacc[((size_t)(b * NPART + p) * 8 + h) * EE + t] * ef[p * 8 + h];
      vbar[h * EE + t] = g2 * (a * invL[h]) + b2;   // softmax weights sum to 1
    }
  }
  __syncthreads();
  {
    const int h = t >> 4, d = t & 15;
    const float* wvp = wv + ((size_t)(layer * HH + h) * EE) * DD + d;
    float a = 0.f;
    for (int e = 0; e < EE; ++e) a += vbar[h * EE + e] * wvp[e * DD];
    attL[t] = a;
  }
  __syncthreads();
  {
    float s = q[b * EE + t] + proj_b[layer * EE + t];
    for (int in = 0; in < EE; ++in) s += attL[in] * proj_w[((size_t)layer * EE + in) * EE + t];
    qbuf[t] = s;
  }
  __syncthreads();
  {
    float v = qbuf[t];
    red[t] = v; __syncthreads();
    for (int s = 64; s > 0; s >>= 1) { if (t < s) red[t] += red[t + s]; __syncthreads(); }
    float mean = red[0] * (1.f / EE); __syncthreads();
    red[t] = (v - mean) * (v - mean); __syncthreads();
    for (int s = 64; s > 0; s >>= 1) { if (t < s) red[t] += red[t + s]; __syncthreads(); }
    float rs = rsqrtf(red[0] * (1.f / EE) + LN_EPS); __syncthreads();
    x4[t] = (v - mean) * rs * ln4_g[layer * EE + t] + ln4_b[layer * EE + t];
  }
  __syncthreads();
  for (int jj = 0; jj < 4; ++jj) {
    const int j = t + jj * EE;
    float s = ffn_b1[layer * 4 * EE + j];
    for (int e = 0; e < EE; ++e) s += x4[e] * ffn_w1[((size_t)layer * EE + e) * (4 * EE) + j];
    hbuf[j] = 0.5f * s * (1.f + erff(s * 0.70710678f));   // exact GELU
  }
  __syncthreads();
  {
    float s = ffn_b2[layer * EE + t];
    for (int j = 0; j < 4 * EE; ++j) s += hbuf[j] * ffn_w2[((size_t)layer * 4 * EE + j) * EE + t];
    q[b * EE + t] = qbuf[t] + s;
  }
}

// ----------------------------------------------------------------- head
__global__ __launch_bounds__(128) void head_out(
    const float* __restrict__ q,
    const float* __restrict__ lnf_g, const float* __restrict__ lnf_b,
    const float* __restrict__ hw, const float* __restrict__ hb,
    float* __restrict__ out)
{
  const int b = blockIdx.x, t = threadIdx.x;
  __shared__ float xf[EE], red[EE];
  float v = q[b * EE + t];
  red[t] = v; __syncthreads();
  for (int s = 64; s > 0; s >>= 1) { if (t < s) red[t] += red[t + s]; __syncthreads(); }
  float mean = red[0] * (1.f / EE); __syncthreads();
  red[t] = (v - mean) * (v - mean); __syncthreads();
  for (int s = 64; s > 0; s >>= 1) { if (t < s) red[t] += red[t + s]; __syncthreads(); }
  float rs = rsqrtf(red[0] * (1.f / EE) + LN_EPS); __syncthreads();
  xf[t] = (v - mean) * rs * lnf_g[t] + lnf_b[t];
  __syncthreads();
  if (t < 37) {
    float s = hb[t];
    for (int e = 0; e < EE; ++e) s += xf[e] * hw[e * 37 + t];
    out[b * 37 + t] = s;
  }
}

extern "C" void kernel_launch(void* const* d_in, const int* in_sizes, int n_in,
                              void* d_out, int out_size, void* d_ws, size_t ws_size,
                              hipStream_t stream)
{
  const float* enc   = (const float*)d_in[0];
  const float* state = (const float*)d_in[1];
  const float* c1w   = (const float*)d_in[2];
  const float* c1b   = (const float*)d_in[3];
  const float* bn1g  = (const float*)d_in[4];
  const float* bn1b  = (const float*)d_in[5];
  const float* bn1m  = (const float*)d_in[6];
  const float* bn1v  = (const float*)d_in[7];
  const float* c2w   = (const float*)d_in[8];
  const float* c2b   = (const float*)d_in[9];
  const float* bn2g  = (const float*)d_in[10];
  const float* bn2b  = (const float*)d_in[11];
  const float* bn2m  = (const float*)d_in[12];
  const float* bn2v  = (const float*)d_in[13];
  const float* lmw   = (const float*)d_in[14];
  const float* lmb   = (const float*)d_in[15];
  const float* wk    = (const float*)d_in[16];
  const float* wq    = (const float*)d_in[17];
  const float* wvw   = (const float*)d_in[18];
  const float* projw = (const float*)d_in[19];
  const float* projb = (const float*)d_in[20];
  const float* ln1g  = (const float*)d_in[21];
  const float* ln1b  = (const float*)d_in[22];
  const float* ln2g  = (const float*)d_in[23];
  const float* ln2b  = (const float*)d_in[24];
  const float* ln3g  = (const float*)d_in[25];
  const float* ln3b  = (const float*)d_in[26];
  const float* ln4g  = (const float*)d_in[27];
  const float* ln4b  = (const float*)d_in[28];
  const float* fw1   = (const float*)d_in[29];
  const float* fb1   = (const float*)d_in[30];
  const float* fw2   = (const float*)d_in[31];
  const float* fb2   = (const float*)d_in[32];
  const float* lnfg  = (const float*)d_in[33];
  const float* lnfb  = (const float*)d_in[34];
  const float* hw    = (const float*)d_in[35];
  const float* hb    = (const float*)d_in[36];

  char* ws = (char*)d_ws;
  float*          q    = (float*)(ws + 0);               // 128*128*4      = 64 KB
  unsigned short* wkqg = (unsigned short*)(ws + 65536);  // 128*16*128*2   = 512 KB
  float*          cq   = (float*)(ws + 589824);          // 128*8*4        = 4 KB
  float*          pm   = (float*)(ws + 593920);          // 128*8*8*4      = 32 KB
  float*          pl   = (float*)(ws + 626688);          // 32 KB
  float*          pacc = (float*)(ws + 659456);          // 128*8*8*128*4  = 4 MB

  conv_head<<<BB, 128, 0, stream>>>(state, c1w, c1b, bn1g, bn1b, bn1m, bn1v,
                                    c2w, c2b, bn2g, bn2b, bn2m, bn2v, lmw, lmb, q);
  for (int layer = 0; layer < NLAYER; ++layer) {
    prep<<<BB, 128, 0, stream>>>(q, ln3g, ln3b, wq, wk, ln1g, ln1b, wkqg, cq, layer);
    attn_pass<<<BB * SPLITN, 128, 0, stream>>>(enc, wkqg, cq, pm, pl, pacc);
    epilogue<<<BB, 128, 0, stream>>>(pm, pl, pacc, ln2g, ln2b, wvw, projw, projb,
                                     ln4g, ln4b, fw1, fb1, fw2, fb2, q, layer);
  }
  head_out<<<BB, 128, 0, stream>>>(q, lnfg, lnfb, hw, hb, (float*)d_out);
}
